// HierarchicalGNNCell_80753975099946
// MI455X (gfx1250) — matrix operand order repaired and checked by reading, not compiled, not run on hardware
//
#include <hip/hip_runtime.h>
#include <hip/hip_bf16.h>
#include <math.h>

typedef __attribute__((ext_vector_type(16))) _Float16 v16h;
typedef __attribute__((ext_vector_type(4)))  _Float16 v4h;
typedef __attribute__((ext_vector_type(8)))  float    v8f;

#define N_NODES  100000
#define N_EDGES  800000
#define N_SUPER  1024
#define N_SEDGES 131072
#define N_BIP    500000
#define LN_EPS   1e-5f
#define WVS      2          // waves per block
#define MT       2          // 16-row M tiles per wave (32 rows/wave, 64 rows/block)

struct MlpP {
  const _Float16* W0; const float* b0; const float* g0; const float* be0;
  const _Float16* W1; const float* b1; const float* g1; const float* be1;
  const _Float16* W2; const float* b2; const float* g2; const float* be2;
};

// Repack f32 W(K x N, row-major) into f16 WMMA-B-fragment order:
// dst[((kt*NT+nt)*32 + lane)*16 + i] = W[(kt*32 + (lane>>4)*16 + i)*N + nt*16 + (lane&15)]
// so each lane's B fragment is one contiguous 32-byte v16h.
__global__ void repack_w_kernel(const float* __restrict__ src,
                                _Float16* __restrict__ dst, int Kdim, int Ndim) {
  int t = blockIdx.x * blockDim.x + threadIdx.x;
  if (t >= Kdim * Ndim) return;
  int i    = t & 15;
  int lane = (t >> 4) & 31;
  int tile = t >> 9;
  int NT = Ndim >> 4;
  int nt = tile % NT;
  int kt = tile / NT;
  int k   = kt * 32 + (lane >> 4) * 16 + i;
  int col = nt * 16 + (lane & 15);
  dst[t] = (_Float16)src[k * Ndim + col];
}

// out[dstIdx[e]*64+c] += (w?w[e]:1) * src[(srcIdx?srcIdx[e]:e)*64+c]
__global__ void scatter_add_kernel(const float* __restrict__ src,
                                   const int* __restrict__ srcIdx,
                                   const int* __restrict__ dstIdx,
                                   const float* __restrict__ w,
                                   float* __restrict__ out, int nE) {
  int t = blockIdx.x * blockDim.x + threadIdx.x;
  int e = t >> 6;
  int c = t & 63;
  if (e >= nE) return;
  int si = srcIdx ? srcIdx[e] : e;
  float v = src[(size_t)si * 64 + c];
  if (w) v *= w[e];
  atomicAdd(&out[(size_t)dstIdx[e] * 64 + c], v);
}

// One wave: 32xN f32 tile = panel(32xK f16, LDS, stride K) * Wp(packed f16) + bias.
// Two 16-row M tiles share every B fragment (one global load -> two WMMAs).
template <int K, int N>
__device__ __forceinline__ void wave_gemm(const _Float16* __restrict__ panel,
                                          const _Float16* __restrict__ Wp,
                                          const float* __restrict__ bias,
                                          float* __restrict__ bufD) {
  constexpr int KT = K / 32, NT = N / 16;
  const int lane = threadIdx.x & 31;
  const int rsel = lane & 15;
  const int hi   = lane >> 4;
  v16h afrag[MT][KT];
#pragma unroll
  for (int m = 0; m < MT; ++m) {
#pragma unroll
    for (int ks = 0; ks < KT; ++ks) {
      const _Float16* ap = panel + (m * 16 + rsel) * K + ks * 32 + hi * 16;
      v16h a;
#pragma unroll
      for (int i = 0; i < 8; ++i) a[i] = ap[i];
#pragma unroll
      for (int i = 0; i < 8; ++i) a[8 + i] = ap[8 + i];
      afrag[m][ks] = a;
    }
  }
  const _Float16* wlane = Wp + (size_t)lane * 16;
#pragma unroll
  for (int nt = 0; nt < NT; ++nt) {
    v8f acc[MT];
#pragma unroll
    for (int m = 0; m < MT; ++m) acc[m] = (v8f){0.f,0.f,0.f,0.f,0.f,0.f,0.f,0.f};
#pragma unroll
    for (int ks = 0; ks < KT; ++ks) {
      const v16h b = *(const v16h*)(wlane + (size_t)(ks * NT + nt) * 512);
#pragma unroll
      for (int m = 0; m < MT; ++m) {
        acc[m] = __builtin_amdgcn_wmma_f32_16x16x32_f16(false, afrag[m][ks], false, b,
                                                        (short)0, acc[m], false, false);
      }
    }
    float bv = bias[nt * 16 + rsel];
#pragma unroll
    for (int m = 0; m < MT; ++m) {
#pragma unroll
      for (int r = 0; r < 8; ++r) {  // D: M = r + 8*hi, N = lane&15
        bufD[(m * 16 + r + hi * 8) * N + nt * 16 + rsel] = acc[m][r] + bv;
      }
    }
  }
}

// LayerNorm + relu over MT*16 rows: lane pair (r, r+16) splits row r (shfl_xor combine).
template <int N>
__device__ __forceinline__ void wave_ln_relu(const float* __restrict__ bufD,
                                             const float* __restrict__ g,
                                             const float* __restrict__ be,
                                             _Float16* __restrict__ panel) {
  const int lane = threadIdx.x & 31;
  const int r = lane & 15, half = lane >> 4;
  const int c0 = half * (N / 2);
#pragma unroll
  for (int m = 0; m < MT; ++m) {
    const float* rowp = bufD + (m * 16 + r) * N + c0;
    float s = 0.f, ss = 0.f;
#pragma unroll
    for (int j = 0; j < N / 2; ++j) { float x = rowp[j]; s += x; ss += x * x; }
    s  += __shfl_xor(s, 16, 32);
    ss += __shfl_xor(ss, 16, 32);
    float mu  = s / (float)N;
    float var = ss / (float)N - mu * mu;
    float inv = rsqrtf(var + LN_EPS);
#pragma unroll
    for (int j = 0; j < N / 2; ++j) {
      float y = (rowp[j] - mu) * inv * g[c0 + j] + be[c0 + j];
      panel[(m * 16 + r) * N + c0 + j] = (_Float16)fmaxf(y, 0.f);
    }
  }
}

// Fused 3-layer MLP (192->128->128->64, LN+act each layer), 32 rows per wave,
// WVS independent waves per block.  Input row i = concat(srcA[iA], srcB[iB], srcC[i]).
// out[i] = act(LN(layer3)) + residual[i].  actMode: 0=relu, 1=tanh.
__global__ void __launch_bounds__(32 * WVS)
mlp_fused_kernel(const float* __restrict__ srcA, const int* __restrict__ idxA,
                 const float* __restrict__ srcB, const int* __restrict__ idxB,
                 const float* __restrict__ srcC,
                 const float* __restrict__ residual,
                 MlpP P, float* __restrict__ out, int R, int actMode) {
  __shared__ __align__(16) _Float16 panel[WVS][32 * 192];  // 24 KB
  __shared__ __align__(16) float    bufD[WVS][32 * 128];   // 32 KB
  __shared__ const float* bases[WVS][32][3];               // 1.5 KB
  const int tid  = threadIdx.x;
  const int w    = tid >> 5;
  const int lane = tid & 31;
  const int base = blockIdx.x * (32 * WVS) + w * 32;

  {  // per-row gather base pointers: one row per lane
    int row = base + lane; if (row >= R) row = R - 1;
    int iA = idxA ? idxA[row] : row;
    int iB = idxB ? idxB[row] : row;
    bases[w][lane][0] = srcA + (size_t)iA * 64;
    bases[w][lane][1] = srcB + (size_t)iB * 64;
    bases[w][lane][2] = srcC + (size_t)row * 64;
  }
  __syncthreads();

  // stage layer-1 panel: 32 rows x 192 cols = 1536 float4 loads, 48 per lane
#pragma unroll
  for (int it = 0; it < 48; ++it) {
    int idx  = lane + 32 * it;
    int row  = idx / 48;
    int q    = idx - row * 48;
    int part = q >> 4, f4 = q & 15;
    const float4 vs = ((const float4*)bases[w][row][part])[f4];
    v4h hv;
    hv[0] = (_Float16)vs.x; hv[1] = (_Float16)vs.y;
    hv[2] = (_Float16)vs.z; hv[3] = (_Float16)vs.w;
    *(v4h*)&panel[w][row * 192 + part * 64 + f4 * 4] = hv;
  }
  __syncthreads();

  wave_gemm<192, 128>(panel[w], P.W0, P.b0, bufD[w]);
  __syncthreads();
  wave_ln_relu<128>(bufD[w], P.g0, P.be0, panel[w]);
  __syncthreads();
  wave_gemm<128, 128>(panel[w], P.W1, P.b1, bufD[w]);
  __syncthreads();
  wave_ln_relu<128>(bufD[w], P.g1, P.be1, panel[w]);
  __syncthreads();
  wave_gemm<128, 64>(panel[w], P.W2, P.b2, bufD[w]);
  __syncthreads();

  {  // final LN + output act + residual (half-row per lane, contiguous stores)
    const int r = lane & 15, half = lane >> 4;
    const int c0 = half * 32;
#pragma unroll
    for (int m = 0; m < MT; ++m) {
      int row = base + m * 16 + r;
      const float* rowp = bufD[w] + (m * 16 + r) * 64 + c0;
      float s = 0.f, ss = 0.f;
#pragma unroll
      for (int j = 0; j < 32; ++j) { float x = rowp[j]; s += x; ss += x * x; }
      s  += __shfl_xor(s, 16, 32);
      ss += __shfl_xor(ss, 16, 32);
      float mu  = s * (1.f / 64.f);
      float var = ss * (1.f / 64.f) - mu * mu;
      float inv = rsqrtf(var + LN_EPS);
      if (row < R) {
        const float* res = residual + (size_t)row * 64 + c0;
        float* op = out + (size_t)row * 64 + c0;
#pragma unroll
        for (int j = 0; j < 32; ++j) {
          float y = (rowp[j] - mu) * inv * P.g2[c0 + j] + P.be2[c0 + j];
          y = actMode ? tanhf(y) : fmaxf(y, 0.f);
          op[j] = y + res[j];
        }
      }
    }
  }
}

extern "C" void kernel_launch(void* const* d_in, const int* in_sizes, int n_in,
                              void* d_out, int out_size, void* d_ws, size_t ws_size,
                              hipStream_t stream) {
  (void)in_sizes; (void)n_in; (void)out_size; (void)ws_size;
  const float* nodes      = (const float*)d_in[0];
  const float* edges      = (const float*)d_in[1];
  const float* supernodes = (const float*)d_in[2];
  const float* superedges = (const float*)d_in[3];
  const int*   graph      = (const int*)d_in[4];
  const int*   bip        = (const int*)d_in[5];
  const float* bw         = (const float*)d_in[6];
  const int*   sgr        = (const int*)d_in[7];
  const float* sw         = (const float*)d_in[8];

  const int* g0  = graph;  const int* g1  = graph + N_EDGES;
  const int* b0i = bip;    const int* b1i = bip + N_BIP;
  const int* s0  = sgr;    const int* s1  = sgr + N_SEDGES;

  float* out       = (float*)d_out;
  float* out_nodes = out;
  float* out_edges = out + (size_t)N_NODES * 64;
  float* out_sn    = out_edges + (size_t)N_EDGES * 64;
  float* out_se    = out_sn + (size_t)N_SUPER * 64;

  float* ws       = (float*)d_ws;
  float* node_msg = ws;                                  // 1024*64
  float* attn_msg = node_msg + (size_t)N_SUPER * 64;     // 1024*64
  float* sn_msg   = attn_msg + (size_t)N_SUPER * 64;     // 100000*64
  float* e_msg    = sn_msg + (size_t)N_NODES * 64;       // 100000*64
  _Float16* wh    = (_Float16*)(e_msg + (size_t)N_NODES * 64);  // 32B-aligned

  // packed f16 weights per MLP: 192*128 + 128*128 + 128*64 = 49152 halves
  const int WOFF[3] = {0, 192 * 128, 192 * 128 + 128 * 128};
  const int KDIM[3] = {192, 128, 128};
  const int NDIM[3] = {128, 128, 64};
  // d_in param order: edge_params@9, node@21, supernode@33, superedge@45,
  // each layer as {W, b, g, beta}.
  MlpP P[4];
  for (int m = 0; m < 4; ++m) {
    int bidx = 9 + 12 * m;
    _Float16* whm = wh + (size_t)m * 49152;
    for (int l = 0; l < 3; ++l) {
      const float* Wf = (const float*)d_in[bidx + 4 * l + 0];
      int cnt = KDIM[l] * NDIM[l];
      repack_w_kernel<<<(cnt + 255) / 256, 256, 0, stream>>>(Wf, whm + WOFF[l],
                                                             KDIM[l], NDIM[l]);
    }
    P[m].W0 = whm + WOFF[0]; P[m].b0 = (const float*)d_in[bidx + 1];
    P[m].g0 = (const float*)d_in[bidx + 2]; P[m].be0 = (const float*)d_in[bidx + 3];
    P[m].W1 = whm + WOFF[1]; P[m].b1 = (const float*)d_in[bidx + 5];
    P[m].g1 = (const float*)d_in[bidx + 6]; P[m].be1 = (const float*)d_in[bidx + 7];
    P[m].W2 = whm + WOFF[2]; P[m].b2 = (const float*)d_in[bidx + 9];
    P[m].g2 = (const float*)d_in[bidx + 10]; P[m].be2 = (const float*)d_in[bidx + 11];
  }
  // P index: 0=edge, 1=node, 2=supernode, 3=superedge

  const int ROWS_PER_BLK = 32 * WVS;

  // --- supernode update ---
  hipMemsetAsync(node_msg, 0, (size_t)2 * N_SUPER * 64 * sizeof(float), stream);
  {
    long long t = (long long)N_BIP * 64;
    scatter_add_kernel<<<(unsigned)((t + 255) / 256), 256, 0, stream>>>(
        nodes, b0i, b1i, bw, node_msg, N_BIP);
    t = (long long)N_SEDGES * 64;
    scatter_add_kernel<<<(unsigned)((t + 255) / 256), 256, 0, stream>>>(
        superedges, nullptr, s1, sw, attn_msg, N_SEDGES);
  }
  mlp_fused_kernel<<<(N_SUPER + ROWS_PER_BLK - 1) / ROWS_PER_BLK, 32 * WVS, 0, stream>>>(
      supernodes, nullptr, attn_msg, nullptr, node_msg, supernodes,
      P[2], out_sn, N_SUPER, 0);

  // --- node update (uses updated supernodes in out_sn) ---
  hipMemsetAsync(sn_msg, 0, (size_t)2 * N_NODES * 64 * sizeof(float), stream);
  {
    long long t = (long long)N_BIP * 64;
    scatter_add_kernel<<<(unsigned)((t + 255) / 256), 256, 0, stream>>>(
        out_sn, b1i, b0i, bw, sn_msg, N_BIP);
    t = (long long)N_EDGES * 64;
    scatter_add_kernel<<<(unsigned)((t + 255) / 256), 256, 0, stream>>>(
        edges, nullptr, g1, nullptr, e_msg, N_EDGES);
  }
  mlp_fused_kernel<<<(N_NODES + ROWS_PER_BLK - 1) / ROWS_PER_BLK, 32 * WVS, 0, stream>>>(
      nodes, nullptr, e_msg, nullptr, sn_msg, nodes,
      P[1], out_nodes, N_NODES, 0);

  // --- superedge update (tanh) ---
  mlp_fused_kernel<<<(N_SEDGES + ROWS_PER_BLK - 1) / ROWS_PER_BLK, 32 * WVS, 0, stream>>>(
      out_sn, s0, out_sn, s1, superedges, superedges,
      P[3], out_se, N_SEDGES, 1);

  // --- edge update (uses updated nodes, tanh) ---
  mlp_fused_kernel<<<(N_EDGES + ROWS_PER_BLK - 1) / ROWS_PER_BLK, 32 * WVS, 0, stream>>>(
      out_nodes, g0, out_nodes, g1, edges, edges,
      P[0], out_edges, N_EDGES, 1);
}
